// CRNLayer_77309411807
// MI455X (gfx1250) — matrix operand order repaired and checked by reading, not compiled, not run on hardware
//
#include <hip/hip_runtime.h>
#include <hip/hip_bf16.h>

// ---------------------------------------------------------------------------
// CDNA5 (gfx1250) implementation: CRN + NetVLAD pipeline via bf16 WMMA.
// Shapes: N=64, C=512, H=W=32, K=64 clusters, P=1024 pixels.
// ---------------------------------------------------------------------------

typedef __attribute__((ext_vector_type(16))) __bf16 v16bf;
typedef __attribute__((ext_vector_type(8)))  float  v8f;

#define NB   64
#define CCH  512
#define HP   16      // pooled H
#define WP   16      // pooled W
#define HF   32
#define WF   32
#define PIX  1024    // HF*WF
#define KCL  64      // clusters
#define CAT_STRIDE 96

__device__ __forceinline__ __bf16 f2bf(float f) {
    unsigned u = __builtin_bit_cast(unsigned, f);
    unsigned r = (u + 0x7FFFu + ((u >> 16) & 1u)) >> 16;
    unsigned short s = (unsigned short)r;
    return __builtin_bit_cast(__bf16, s);
}

__device__ __forceinline__ v16bf zero_frag() {
    __bf16 z = __builtin_bit_cast(__bf16, (unsigned short)0);
    v16bf v;
#pragma unroll
    for (int e = 0; e < 16; ++e) v[e] = z;
    return v;
}

// A fragment (16x32 bf16, M x K). Row m = lane&15, row stride = ldr elements.
// Layout: lanes 0-15 hold K 0..15, lanes 16-31 hold K 16..31 (matches B).
__device__ __forceinline__ v16bf load_frag_a(const __bf16* __restrict__ p,
                                             int lane, int ldr) {
    int m  = lane & 15;
    int kb = (lane >> 4) << 4;          // 0 or 16
    const __bf16* r = p + (size_t)m * ldr;
    v16bf v;
#pragma unroll
    for (int e = 0; e < 16; ++e) v[e]    = r[kb + e];
    return v;
}

// B fragment (32x16 bf16, K x N). Column n = lane&15 at p + n*ldc,
// K rows contiguous; lanes 0-15 take K 0..15, lanes 16-31 take K 16..31.
__device__ __forceinline__ v16bf load_frag_b(const __bf16* __restrict__ p,
                                             int lane, int ldc) {
    int c  = lane & 15;
    int kb = (lane >> 4) << 4;          // 0 or 16
    const __bf16* r = p + (size_t)c * ldc + kb;
    v16bf v;
#pragma unroll
    for (int e = 0; e < 16; ++e) v[e] = r[e];
    return v;
}

__device__ __forceinline__ v8f wmma_bf16(v16bf a, v16bf b, v8f c) {
    return __builtin_amdgcn_wmma_f32_16x16x32_bf16(
        /*neg_a=*/false, a, /*neg_b=*/false, b,
        /*c_mod=*/(short)0, c, /*reuse_a=*/false, /*reuse_b=*/false);
}

// ---------------------------------------------------------------------------
// 1) 2x2 average pool, fp32 NCHW -> bf16 channel-last [n][ho][wo][C]
// ---------------------------------------------------------------------------
__global__ void avgpool_bf16_k(const float* __restrict__ x, __bf16* __restrict__ m) {
    size_t t = (size_t)blockIdx.x * blockDim.x + threadIdx.x;
    if (t >= (size_t)NB * HP * WP * (CCH / 4)) return;
    int cg = (int)(t & 127) * 4;        // channel group of 4
    size_t pix = t >> 7;                // n*256 + ho*16 + wo
    int wo = (int)(pix & 15);
    int ho = (int)((pix >> 4) & 15);
    size_t n = pix >> 8;
    const float* px = x + (((n * CCH + cg) * HF) + ho * 2) * WF + wo * 2;
    __bf16* pm = m + pix * CCH + cg;
#pragma unroll
    for (int i = 0; i < 4; ++i) {
        const float* q = px + (size_t)i * (HF * WF);
        pm[i] = f2bf(0.25f * (q[0] + q[1] + q[WF] + q[WF + 1]));
    }
}

// ---------------------------------------------------------------------------
// 2) Weight transform: OIHW fp32 -> [kh][kw][cout pad32][cin] bf16
// ---------------------------------------------------------------------------
__global__ void wtrans_conv_k(const float* __restrict__ w, __bf16* __restrict__ wt,
                              int Cout, int KS) {
    int idx = blockIdx.x * blockDim.x + threadIdx.x;
    int total = KS * KS * 32 * CCH;
    if (idx >= total) return;
    int ci = idx & (CCH - 1);
    int co = (idx >> 9) & 31;
    int kk = idx >> 14;                 // kh*KS + kw
    float v = 0.f;
    if (co < Cout) v = w[((size_t)co * CCH + ci) * (KS * KS) + kk];
    wt[idx] = f2bf(v);
}

__global__ void wtrans_flat_k(const float* __restrict__ w, __bf16* __restrict__ wt,
                              int total) {
    int idx = blockIdx.x * blockDim.x + threadIdx.x;
    if (idx < total) wt[idx] = f2bf(w[idx]);
}

// ---------------------------------------------------------------------------
// 3) Implicit-GEMM convolution with WMMA bf16.
//    Tile: 16 out-channels x 16 pixels (one pooled row) per wave.
//    grid = (16 rows, 64 images), block = 64 (2 waves = 2 out-channel tiles)
// ---------------------------------------------------------------------------
template <int KS, int PAD>
__global__ void conv_wmma_k(const __bf16* __restrict__ m,   // [n][h][w][C]
                            const __bf16* __restrict__ wt,  // [KS][KS][32][C]
                            const float* __restrict__ bias,
                            float* __restrict__ cat,        // [n][h][w][96]
                            int coutValid, int chOff) {
    int lane = threadIdx.x & 31;
    int mt   = threadIdx.x >> 5;        // out-channel tile 0/1
    int h = blockIdx.x;
    int n = blockIdx.y;
    int wpix = lane & 15;
    int kb = (lane >> 4) << 4;          // B-fragment K base within 32-chunk
    v8f acc = {};
    for (int kh = 0; kh < KS; ++kh) {
        int ih = h + kh - PAD;
        if (ih < 0 || ih >= HP) continue;
        for (int kw = 0; kw < KS; ++kw) {
            int iw = wpix + kw - PAD;
            bool okB = (iw >= 0 && iw < WP);
            const __bf16* bbase =
                m + ((((size_t)n * HP + ih) * WP) + (okB ? iw : 0)) * CCH;
            const __bf16* abase = wt + (((size_t)(kh * KS + kw)) * 32 + mt * 16) * CCH;
            for (int cb = 0; cb < CCH; cb += 32) {
                v16bf a = load_frag_a(abase + cb, lane, CCH);
                v16bf b;
                if (okB) {
                    const __bf16* r = bbase + cb + kb;
#pragma unroll
                    for (int e = 0; e < 16; ++e) b[e] = r[e];
                } else {
                    b = zero_frag();
                }
                acc = wmma_bf16(a, b, acc);
            }
        }
    }
    int rb = mt * 16 + ((lane >> 4) << 3);
#pragma unroll
    for (int r = 0; r < 8; ++r) {
        int co = rb + r;
        if (co < coutValid) {
            float v = acc[r] + bias[co];
            cat[((((size_t)n * HP + h) * WP) + wpix) * CAT_STRIDE + chOff + co] =
                fmaxf(v, 0.f);
        }
    }
}

// ---------------------------------------------------------------------------
// 4) 1x1 conv over concat(84) + bias -> mmS [n][16][16]
// ---------------------------------------------------------------------------
__global__ void crn_reduce_k(const float* __restrict__ cat,
                             const float* __restrict__ wa,
                             const float* __restrict__ ba,
                             float* __restrict__ mmS) {
    int idx = blockIdx.x * blockDim.x + threadIdx.x;
    if (idx >= NB * HP * WP) return;
    const float* p = cat + (size_t)idx * CAT_STRIDE;
    float s = ba[0];
#pragma unroll 4
    for (int j = 0; j < 84; ++j) s += wa[j] * p[j];
    mmS[idx] = s;
}

// ---------------------------------------------------------------------------
// 5) Bilinear 2x upsample (half-pixel, edge clamp) -> mm [n][32][32]
// ---------------------------------------------------------------------------
__global__ void upsample2x_k(const float* __restrict__ mmS, float* __restrict__ mm) {
    int idx = blockIdx.x * blockDim.x + threadIdx.x;
    if (idx >= NB * PIX) return;
    int j = idx & 31, i = (idx >> 5) & 31, n = idx >> 10;
    float si = 0.5f * i - 0.25f, sj = 0.5f * j - 0.25f;
    int i0 = (int)floorf(si), j0 = (int)floorf(sj);
    float fi = si - (float)i0, fj = sj - (float)j0;
    int i0c = i0 < 0 ? 0 : (i0 > 15 ? 15 : i0);
    int i1 = i0 + 1; int i1c = i1 < 0 ? 0 : (i1 > 15 ? 15 : i1);
    int j0c = j0 < 0 ? 0 : (j0 > 15 ? 15 : j0);
    int j1 = j0 + 1; int j1c = j1 < 0 ? 0 : (j1 > 15 ? 15 : j1);
    const float* b = mmS + (size_t)n * (HP * WP);
    float v = (1.f - fi) * ((1.f - fj) * b[i0c * WP + j0c] + fj * b[i0c * WP + j1c])
            +        fi  * ((1.f - fj) * b[i1c * WP + j0c] + fj * b[i1c * WP + j1c]);
    mm[idx] = v;
}

// ---------------------------------------------------------------------------
// 6) Per-pixel inverse channel L2 norm of x
// ---------------------------------------------------------------------------
__global__ void invnorm_k(const float* __restrict__ x, float* __restrict__ invn) {
    int idx = blockIdx.x * blockDim.x + threadIdx.x;
    if (idx >= NB * PIX) return;
    int p = idx & (PIX - 1);
    size_t n = (size_t)(idx >> 10);
    const float* q = x + n * CCH * PIX + p;
    float s = 0.f;
    for (int c = 0; c < CCH; ++c) { float v = q[(size_t)c << 10]; s += v * v; }
    invn[idx] = 1.f / fmaxf(sqrtf(s), 1e-12f);
}

// ---------------------------------------------------------------------------
// 7) xnT bf16 [n][p][C] via LDS transpose (B operand for GEMM1)
// ---------------------------------------------------------------------------
__global__ void xnT_k(const float* __restrict__ x, const float* __restrict__ invn,
                      __bf16* __restrict__ xnT) {
    __shared__ __bf16 tile[32][33];
    int n = blockIdx.z;
    int c0 = blockIdx.y * 32, p0 = blockIdx.x * 32;
    int tx = threadIdx.x, ty = threadIdx.y;
    for (int i = ty; i < 32; i += 8) {
        float v = x[(((size_t)n * CCH) + c0 + i) * PIX + p0 + tx]
                * invn[n * PIX + p0 + tx];
        tile[i][tx] = f2bf(v);
    }
    __syncthreads();
    for (int i = ty; i < 32; i += 8) {
        xnT[(((size_t)n * PIX) + p0 + i) * CCH + c0 + tx] = tile[tx][i];
    }
}

// ---------------------------------------------------------------------------
// 8) GEMM1: sa[n] = wvt(64x512) * xnT[n](512x1024)  (K-dim = channels)
//    grid = (32 pixel-tile-pairs, 64 images), block 256 = 8 waves
// ---------------------------------------------------------------------------
__global__ void gemm_sa_k(const __bf16* __restrict__ wvt,  // [64][512]
                          const __bf16* __restrict__ xnT,  // [n][1024][512]
                          float* __restrict__ sa) {        // [n][64][1024]
    int lane = threadIdx.x & 31, w = threadIdx.x >> 5;
    int mt = w & 3;
    int nt = blockIdx.x * 2 + (w >> 2);
    int n  = blockIdx.y;
    const __bf16* A = wvt + (size_t)mt * 16 * CCH;
    const __bf16* B = xnT + (((size_t)n * PIX) + nt * 16) * CCH;
    v8f acc = {};
    for (int cb = 0; cb < CCH; cb += 32) {
        v16bf a = load_frag_a(A + cb, lane, CCH);
        v16bf b = load_frag_b(B + cb, lane, CCH);
        acc = wmma_bf16(a, b, acc);
    }
    int col = nt * 16 + (lane & 15);
    int rb  = mt * 16 + ((lane >> 4) << 3);
#pragma unroll
    for (int r = 0; r < 8; ++r)
        sa[(((size_t)n * KCL) + rb + r) * PIX + col] = acc[r];
}

// ---------------------------------------------------------------------------
// 9) Softmax over clusters * mm -> a (bf16) and asum[n][k] (atomic)
// ---------------------------------------------------------------------------
__global__ void softmax_k(const float* __restrict__ sa, const float* __restrict__ mm,
                          __bf16* __restrict__ aB, float* __restrict__ asum) {
    int tid = threadIdx.x;
    int n = blockIdx.y;
    int p = blockIdx.x * 256 + tid;
    const float* s = sa + ((size_t)n * KCL) * PIX + p;
    float mx = -1e30f;
    for (int k = 0; k < KCL; ++k) mx = fmaxf(mx, s[(size_t)k << 10]);
    float den = 0.f;
    for (int k = 0; k < KCL; ++k) den += __expf(s[(size_t)k << 10] - mx);
    float scale = mm[n * PIX + p] / den;
    int lane = tid & 31;
    for (int k = 0; k < KCL; ++k) {
        float v = __expf(s[(size_t)k << 10] - mx) * scale;
        aB[(((size_t)n * KCL) + k) * PIX + p] = f2bf(v);
        float r = v;
#pragma unroll
        for (int off = 16; off > 0; off >>= 1) r += __shfl_down(r, off, 32);
        if (lane == 0) atomicAdd(&asum[n * KCL + k], r);
    }
}

// ---------------------------------------------------------------------------
// 10) GEMM2: vlad[n] = a[n](64x1024) * xn[n]^T(1024x512) - asum*centroids
//     B operand built on the fly from fp32 x * invn (contiguous pixels).
//     grid = (16 channel-tile-pairs, 64 images), block 256 = 8 waves
// ---------------------------------------------------------------------------
__global__ void gemm_vlad_k(const __bf16* __restrict__ aB,   // [n][64][1024]
                            const float* __restrict__ x,     // [n][C][1024]
                            const float* __restrict__ invn,  // [n][1024]
                            const float* __restrict__ asum,  // [n][64]
                            const float* __restrict__ cent,  // [64][512]
                            float* __restrict__ vlad) {      // [n][64][512]
    int lane = threadIdx.x & 31, w = threadIdx.x >> 5;
    int mt = w & 3;
    int nt = blockIdx.x * 2 + (w >> 2);
    int n  = blockIdx.y;
    int ch = nt * 16 + (lane & 15);
    const float* Bx = x + (((size_t)n * CCH) + ch) * PIX;
    const float* iv = invn + (size_t)n * PIX;
    const __bf16* A = aB + (((size_t)n * KCL) + mt * 16) * PIX;
    v8f acc = {};
    for (int pb = 0; pb < PIX; pb += 32) {
        v16bf a = load_frag_a(A + pb, lane, PIX);
        int kbase = pb + ((lane >> 4) << 4);
        v16bf b;
#pragma unroll
        for (int e = 0; e < 16; ++e) b[e] = f2bf(Bx[kbase + e] * iv[kbase + e]);
        acc = wmma_bf16(a, b, acc);
    }
    int rb = mt * 16 + ((lane >> 4) << 3);
#pragma unroll
    for (int r = 0; r < 8; ++r) {
        int k = rb + r;
        float v = acc[r] - asum[n * KCL + k] * cent[(size_t)k * CCH + ch];
        vlad[(((size_t)n * KCL) + k) * CCH + ch] = v;
    }
}

// ---------------------------------------------------------------------------
// 11) Final normalization chain
// ---------------------------------------------------------------------------
__global__ void rownorm_k(const float* __restrict__ vlad, float* __restrict__ rowsq) {
    int row = blockIdx.x;               // n*64 + k
    const float* p = vlad + (size_t)row * CCH;
    int t = threadIdx.x;                // 64 threads
    float s = 0.f;
    for (int i = t; i < CCH; i += 64) { float v = p[i]; s += v * v; }
    __shared__ float red[64];
    red[t] = s;
    __syncthreads();
    if (t < 32) red[t] += red[t + 32];
    __syncthreads();
    if (t == 0) {
        float tot = 0.f;
        for (int i = 0; i < 32; ++i) tot += red[i];
        rowsq[row] = tot;
    }
}

__global__ void totnorm_k(const float* __restrict__ rowsq, float* __restrict__ rowinv,
                          float* __restrict__ totinv) {
    int n = blockIdx.x;
    int k = threadIdx.x;                // 64 threads
    float sq = rowsq[n * KCL + k];
    float denom = fmaxf(sqrtf(sq), 1e-12f);
    float ri = 1.f / denom;
    rowinv[n * KCL + k] = ri;
    __shared__ float red[64];
    red[k] = sq * ri * ri;
    __syncthreads();
    if (k == 0) {
        float s = 0.f;
        for (int i = 0; i < KCL; ++i) s += red[i];
        totinv[n] = 1.f / fmaxf(sqrtf(s), 1e-12f);
    }
}

__global__ void finalize_k(const float* __restrict__ vlad,
                           const float* __restrict__ rowinv,
                           const float* __restrict__ totinv,
                           float* __restrict__ out) {
    size_t idx = (size_t)blockIdx.x * 256 + threadIdx.x;
    if (idx >= (size_t)NB * KCL * CCH) return;
    size_t n = idx >> 15;               // / 32768
    int kc = (int)(idx & 32767);
    int k = kc >> 9;
    out[idx] = vlad[idx] * rowinv[n * KCL + k] * totinv[n];
}

// ---------------------------------------------------------------------------
// Host launch
// ---------------------------------------------------------------------------
extern "C" void kernel_launch(void* const* d_in, const int* in_sizes, int n_in,
                              void* d_out, int out_size, void* d_ws, size_t ws_size,
                              hipStream_t stream) {
    (void)in_sizes; (void)n_in; (void)out_size; (void)ws_size;
    const float* x    = (const float*)d_in[0];
    const float* w1   = (const float*)d_in[1];
    const float* b1   = (const float*)d_in[2];
    const float* w2   = (const float*)d_in[3];
    const float* b2   = (const float*)d_in[4];
    const float* w3   = (const float*)d_in[5];
    const float* b3   = (const float*)d_in[6];
    const float* wa   = (const float*)d_in[7];
    const float* ba   = (const float*)d_in[8];
    const float* wv   = (const float*)d_in[9];
    const float* cent = (const float*)d_in[10];
    float* out = (float*)d_out;

    char* ws = (char*)d_ws;
    auto carve = [&](size_t bytes) -> char* {
        char* p = ws;
        ws += (bytes + 255) & ~(size_t)255;
        return p;
    };

    __bf16* mB    = (__bf16*)carve((size_t)NB * HP * WP * CCH * 2);       // 16.8 MB
    __bf16* w1t   = (__bf16*)carve((size_t)9  * 32 * CCH * 2);
    __bf16* w2t   = (__bf16*)carve((size_t)25 * 32 * CCH * 2);
    __bf16* w3t   = (__bf16*)carve((size_t)49 * 32 * CCH * 2);
    __bf16* wvt   = (__bf16*)carve((size_t)KCL * CCH * 2);
    float*  cat   = (float*) carve((size_t)NB * HP * WP * CAT_STRIDE * 4); // 6.3 MB
    float*  mmS   = (float*) carve((size_t)NB * HP * WP * 4);
    float*  mm    = (float*) carve((size_t)NB * PIX * 4);
    float*  invn  = (float*) carve((size_t)NB * PIX * 4);
    __bf16* xnT   = (__bf16*)carve((size_t)NB * PIX * CCH * 2);           // 67 MB
    float*  sa    = (float*) carve((size_t)NB * KCL * PIX * 4);           // 16.8 MB
    __bf16* aB    = (__bf16*)carve((size_t)NB * KCL * PIX * 2);           // 8.4 MB
    float*  asum  = (float*) carve((size_t)NB * KCL * 4);
    float*  vlad  = (float*) carve((size_t)NB * KCL * CCH * 4);           // 8.4 MB
    float*  rowsq = (float*) carve((size_t)NB * KCL * 4);
    float*  rowiv = (float*) carve((size_t)NB * KCL * 4);
    float*  totiv = (float*) carve((size_t)NB * 4);

    // 1) pool + bf16
    avgpool_bf16_k<<<dim3((NB * HP * WP * (CCH / 4) + 255) / 256), 256, 0, stream>>>(x, mB);
    // 2) weight transforms
    wtrans_conv_k<<<dim3((9  * 32 * CCH + 255) / 256), 256, 0, stream>>>(w1, w1t, 32, 3);
    wtrans_conv_k<<<dim3((25 * 32 * CCH + 255) / 256), 256, 0, stream>>>(w2, w2t, 32, 5);
    wtrans_conv_k<<<dim3((49 * 32 * CCH + 255) / 256), 256, 0, stream>>>(w3, w3t, 20, 7);
    wtrans_flat_k<<<dim3((KCL * CCH + 255) / 256), 256, 0, stream>>>(wv, wvt, KCL * CCH);
    // 3) convs (implicit GEMM, WMMA bf16)
    conv_wmma_k<3, 1><<<dim3(HP, NB), 64, 0, stream>>>(mB, w1t, b1, cat, 32, 0);
    conv_wmma_k<5, 2><<<dim3(HP, NB), 64, 0, stream>>>(mB, w2t, b2, cat, 32, 32);
    conv_wmma_k<7, 3><<<dim3(HP, NB), 64, 0, stream>>>(mB, w3t, b3, cat, 20, 64);
    // 4) 1x1 conv + 5) upsample
    crn_reduce_k<<<dim3((NB * HP * WP + 255) / 256), 256, 0, stream>>>(cat, wa, ba, mmS);
    upsample2x_k<<<dim3((NB * PIX + 255) / 256), 256, 0, stream>>>(mmS, mm);
    // 6) inv norms, 7) xnT transpose
    invnorm_k<<<dim3((NB * PIX + 255) / 256), 256, 0, stream>>>(x, invn);
    xnT_k<<<dim3(PIX / 32, CCH / 32, NB), dim3(32, 8), 0, stream>>>(x, invn, xnT);
    // 8) GEMM1: soft-assignment logits
    gemm_sa_k<<<dim3(32, NB), 256, 0, stream>>>(wvt, xnT, sa);
    // 9) softmax * mm  (asum accumulated atomically)
    hipMemsetAsync(asum, 0, (size_t)NB * KCL * 4, stream);
    softmax_k<<<dim3(PIX / 256, NB), 256, 0, stream>>>(sa, mm, aB, asum);
    // 10) GEMM2: VLAD aggregation with fused centroid subtraction
    gemm_vlad_k<<<dim3(16, NB), 256, 0, stream>>>(aB, x, invn, asum, cent, vlad);
    // 11) intra-norm + global norm + write
    rownorm_k<<<dim3(NB * KCL), 64, 0, stream>>>(vlad, rowsq);
    totnorm_k<<<dim3(NB), 64, 0, stream>>>(rowsq, rowiv, totiv);
    finalize_k<<<dim3((NB * KCL * CCH + 255) / 256), 256, 0, stream>>>(vlad, rowiv, totiv, out);
}
